// MultiHeadAttention_62380105007909
// MI455X (gfx1250) — matrix operand order repaired and checked by reading, not compiled, not run on hardware
//
#include <hip/hip_runtime.h>

// ---------------------------------------------------------------------------
// MultiHeadAttention for B=4,T=1024,C=1024,H=16,D=64 on gfx1250 (MI455X).
// All matmuls via v_wmma_f32_16x16x32_bf16 (f32 accumulate); global->LDS
// staging via GLOBAL_LOAD_ASYNC_TO_LDS_B128 (ASYNCcnt) where layout permits.
// Workspace (ushort bf16 bits), ~48 MB total:
//   x_bf[4096*1024] | wqkv[1024*3072] | wp_bf[1024*1024]
//   qkv[3][B,H,T,D] | attn[4096*1024]
// ---------------------------------------------------------------------------

typedef __attribute__((ext_vector_type(16))) __bf16        v16bf;
typedef __attribute__((ext_vector_type(8)))  float         v8f;
typedef __attribute__((ext_vector_type(4)))  unsigned int  u32x4;

union Frag16 {
    v16bf          v;
    unsigned short u[16];
};

__device__ __forceinline__ void cp16(void* d, const void* s) {
    *reinterpret_cast<u32x4*>(d) = *reinterpret_cast<const u32x4*>(s);
}

// async global->LDS, 16 bytes per lane; tracked by ASYNCcnt
__device__ __forceinline__ void async_ld16(const unsigned short* lds_ptr,
                                           const unsigned short* gptr) {
    unsigned lds_off = (unsigned)(unsigned long long)lds_ptr;  // ADDR[31:0] = LDS offset
    asm volatile("global_load_async_to_lds_b128 %0, %1, off"
                 :: "v"(lds_off), "v"(gptr)
                 : "memory");
}
__device__ __forceinline__ void wait_async() {
    asm volatile("s_wait_asynccnt 0x0" ::: "memory");
}

__device__ __forceinline__ unsigned short f2bf(float f) {
    unsigned u = __float_as_uint(f);
    unsigned r = u + 0x7FFFu + ((u >> 16) & 1u);   // round-to-nearest-even
    return (unsigned short)(r >> 16);
}

__device__ __forceinline__ v8f wmma_bf16(const Frag16& a, const Frag16& b, v8f c) {
    return __builtin_amdgcn_wmma_f32_16x16x32_bf16(
        false, a.v, false, b.v, (short)0, c, false, false);
}

// ---------------------------------------------------------------------------
// Conversion / packing kernels
// ---------------------------------------------------------------------------
__global__ void cvt_f32_bf16(const float* __restrict__ in,
                             unsigned short* __restrict__ out, int n) {
    int i = blockIdx.x * 256 + threadIdx.x;
    if (i < n) out[i] = f2bf(in[i]);
}

// Wq/Wk/Wv are (H=16, C=1024, D=64). Pack into [c][sec*1024 + h*64 + d].
__global__ void pack_wqkv(const float* __restrict__ Wq,
                          const float* __restrict__ Wk,
                          const float* __restrict__ Wv,
                          unsigned short* __restrict__ out) {
    int i = blockIdx.x * 256 + threadIdx.x;   // over 1024*3072
    int c   = i / 3072;
    int rr  = i % 3072;
    int sec = rr >> 10;
    int hd  = rr & 1023;
    int h   = hd >> 6;
    int d   = hd & 63;
    const float* src = (sec == 0) ? Wq : (sec == 1) ? Wk : Wv;
    out[i] = f2bf(src[((size_t)h * 1024 + c) * 64 + d]);
}

// ---------------------------------------------------------------------------
// bf16 GEMM: Y[M,N] = A[M,K] * W[K,N].  256 thr (8 waves), tile 128x128, BK=32.
// Each wave: 16 rows x 128 cols -> 8 C tiles, 8 WMMAs per k-step, 1 A-frag.
// MODE 0: scatter-store bf16 into qkv[3][B,H,T,D]   (N = 3072)
// MODE 1: f32 store Y + bias into outf              (N = 1024)
// ---------------------------------------------------------------------------
template <int MODE>
__global__ __launch_bounds__(256) void gemm_bf16(
    const unsigned short* __restrict__ A,
    const unsigned short* __restrict__ W,
    int M, int N, int K,
    unsigned short* __restrict__ qkv,
    float* __restrict__ outf,
    const float* __restrict__ bias)
{
    __shared__ unsigned short Xs[128][40];   // [m][k]  (+8 pad, 16B-aligned rows)
    __shared__ unsigned short Ws[128][40];   // [n][k]  transposed

    const int tid  = threadIdx.x;
    const int lane = tid & 31;
    const int wv   = tid >> 5;
    const int hlf  = lane >> 4;
    const int mrow = lane & 15;
    const int kA   = hlf * 8;    // A frag: K = kA + i (i<8), kA + 16 + (i-8)
    const int kB   = hlf * 8;    // B frag: K = kB + i (i<8), kB + 16 + (i-8)
    const int m0   = blockIdx.x * 128;
    const int n0   = blockIdx.y * 128;

    v8f acc[8] = {};

    for (int k0 = 0; k0 < K; k0 += 32) {
        // async stage X tile 128x32 (512 chunks of 8 halfs, 2 per thread)
        for (int ch = tid; ch < 512; ch += 256) {
            int r = ch >> 2, c4 = ch & 3;
            async_ld16(&Xs[r][c4 * 8], &A[(size_t)(m0 + r) * K + k0 + c4 * 8]);
        }
        // stage W tile 32x128, transposed into Ws[n][k]
        for (int e = tid; e < 4096; e += 256) {
            int n = e & 127, kk = e >> 7;
            Ws[n][kk] = W[(size_t)(k0 + kk) * N + n0 + n];
        }
        wait_async();
        __syncthreads();

        Frag16 a;
        const unsigned short* ar = &Xs[wv * 16 + mrow][0];
        cp16(a.u + 0, ar + kA);
        cp16(a.u + 8, ar + kA + 16);
#pragma unroll
        for (int nt = 0; nt < 8; ++nt) {
            Frag16 b;
            const unsigned short* br = &Ws[nt * 16 + mrow][0];
            cp16(b.u + 0, br + kB);
            cp16(b.u + 8, br + kB + 16);
            acc[nt] = wmma_bf16(a, b, acc[nt]);
        }
        __syncthreads();
    }

#pragma unroll
    for (int nt = 0; nt < 8; ++nt) {
#pragma unroll
        for (int r = 0; r < 8; ++r) {
            int m = m0 + wv * 16 + r + hlf * 8;
            int n = n0 + nt * 16 + mrow;
            float v = acc[nt][r];
            if (MODE == 0) {
                int sec = n >> 10;
                int rr  = n & 1023;
                int h   = rr >> 6, d = rr & 63;
                int b   = m >> 10, t = m & 1023;
                qkv[(size_t)sec * 4194304 +
                    (((size_t)b * 16 + h) * 1024 + t) * 64 + d] = f2bf(v);
            } else {
                outf[(size_t)m * N + n] = v + bias[n];
            }
        }
    }
}

// ---------------------------------------------------------------------------
// Flash attention: one block per (qblock, h, b). 128 thr = 4 waves.
// Each wave owns 16 query rows; block processes 64 q-rows x 64 k-cols tiles.
// ---------------------------------------------------------------------------
__global__ __launch_bounds__(128) void attn_kernel(
    const unsigned short* __restrict__ Qg,   // [B,H,T,D] bf16
    const unsigned short* __restrict__ Kg,
    const unsigned short* __restrict__ Vg,
    unsigned short* __restrict__ Og,         // [B,T,H*D] bf16
    float scale)
{
    __shared__ unsigned short Qs [64][72];       // [t][d]
    __shared__ unsigned short Ks [64][72];       // [s][d]
    __shared__ unsigned short Vts[64][72];       // [d][s]  (transposed)
    __shared__ unsigned short Ps [4][16][72];    // per-wave P tile [t][s]

    const int qb   = blockIdx.x;       // 0..15
    const int h    = blockIdx.y;
    const int b    = blockIdx.z;
    const int tid  = threadIdx.x;
    const int lane = tid & 31;
    const int wv   = tid >> 5;
    const int hlf  = lane >> 4;
    const int mrow = lane & 15;
    const int kA   = hlf * 8;
    const int kB   = hlf * 8;

    const size_t bh = ((size_t)b * 16 + h) * 1024 * 64;
    const int q0 = qb * 64;

    // async stage Q block (64x64 halfs = 512 chunks, 4 per thread)
    for (int ch = tid; ch < 512; ch += 128) {
        int r = ch >> 3, c8 = ch & 7;
        async_ld16(&Qs[r][c8 * 8], &Qg[bh + (size_t)(q0 + r) * 64 + c8 * 8]);
    }

    v8f   o[4] = {};
    float m_run[8], l_run[8];
#pragma unroll
    for (int r = 0; r < 8; ++r) { m_run[r] = -1e30f; l_run[r] = 0.0f; }

    wait_async();
    __syncthreads();

    for (int s0 = 0; s0 <= q0 + 63; s0 += 64) {
        // async stage K (direct); scalar stage V (transposed)
        for (int ch = tid; ch < 512; ch += 128) {
            int r = ch >> 3, c8 = ch & 7;
            async_ld16(&Ks[r][c8 * 8], &Kg[bh + (size_t)(s0 + r) * 64 + c8 * 8]);
        }
        for (int e = tid; e < 4096; e += 128) {
            int r = e >> 6, c = e & 63;
            Vts[c][r] = Vg[bh + (size_t)(s0 + r) * 64 + c];
        }
        wait_async();
        __syncthreads();

        // ---- S = Q * K^T  (4 tiles of 16x16, K-dim = 64 = 2 WMMAs) ----
        Frag16 aq0, aq1;
        const unsigned short* qr = &Qs[wv * 16 + mrow][0];
        cp16(aq0.u + 0, qr + kA);        cp16(aq0.u + 8, qr + kA + 16);
        cp16(aq1.u + 0, qr + 32 + kA);   cp16(aq1.u + 8, qr + 32 + kA + 16);

        v8f st[4];
#pragma unroll
        for (int nt = 0; nt < 4; ++nt) {
            Frag16 bk0, bk1;
            const unsigned short* kr = &Ks[nt * 16 + mrow][0];
            cp16(bk0.u + 0, kr + kB);        cp16(bk0.u + 8, kr + kB + 16);
            cp16(bk1.u + 0, kr + 32 + kB);   cp16(bk1.u + 8, kr + 32 + kB + 16);
            v8f c = {};
            c = wmma_bf16(aq0, bk0, c);
            c = wmma_bf16(aq1, bk1, c);
            st[nt] = c;
        }

        // ---- scale + causal mask (diagonal block only) ----
        const bool diag = (s0 == q0);
#pragma unroll
        for (int nt = 0; nt < 4; ++nt) {
#pragma unroll
            for (int r = 0; r < 8; ++r) {
                float v = st[nt][r] * scale;
                if (diag) {
                    int trow = wv * 16 + r + hlf * 8;
                    int scol = nt * 16 + mrow;
                    if (scol > trow) v = -1e30f;
                }
                st[nt][r] = v;
            }
        }

        // ---- online softmax (row reductions across 16-lane halves) ----
        float corr[8];
#pragma unroll
        for (int r = 0; r < 8; ++r) {
            float mx = fmaxf(fmaxf(st[0][r], st[1][r]), fmaxf(st[2][r], st[3][r]));
            mx = fmaxf(mx, __shfl_xor(mx, 1, 32));
            mx = fmaxf(mx, __shfl_xor(mx, 2, 32));
            mx = fmaxf(mx, __shfl_xor(mx, 4, 32));
            mx = fmaxf(mx, __shfl_xor(mx, 8, 32));
            float mn = fmaxf(m_run[r], mx);
            corr[r]  = __expf(m_run[r] - mn);
            m_run[r] = mn;
            float rs = 0.0f;
#pragma unroll
            for (int nt = 0; nt < 4; ++nt) {
                float p = __expf(st[nt][r] - mn);
                st[nt][r] = p;
                rs += p;
            }
            rs += __shfl_xor(rs, 1, 32);
            rs += __shfl_xor(rs, 2, 32);
            rs += __shfl_xor(rs, 4, 32);
            rs += __shfl_xor(rs, 8, 32);
            l_run[r] = l_run[r] * corr[r] + rs;
        }

        // rescale running output, spill P (C-layout -> A-layout via LDS)
#pragma unroll
        for (int dt = 0; dt < 4; ++dt)
#pragma unroll
            for (int r = 0; r < 8; ++r) o[dt][r] *= corr[r];

#pragma unroll
        for (int nt = 0; nt < 4; ++nt)
#pragma unroll
            for (int r = 0; r < 8; ++r)
                Ps[wv][r + hlf * 8][nt * 16 + mrow] = f2bf(st[nt][r]);
        __syncthreads();

        // ---- O += P * V  (4 tiles along D, K-dim = 64 = 2 WMMAs) ----
        Frag16 ap0, ap1;
        const unsigned short* pr = &Ps[wv][mrow][0];
        cp16(ap0.u + 0, pr + kA);        cp16(ap0.u + 8, pr + kA + 16);
        cp16(ap1.u + 0, pr + 32 + kA);   cp16(ap1.u + 8, pr + 32 + kA + 16);
#pragma unroll
        for (int dt = 0; dt < 4; ++dt) {
            Frag16 bv0, bv1;
            const unsigned short* vr = &Vts[dt * 16 + mrow][0];
            cp16(bv0.u + 0, vr + kB);        cp16(bv0.u + 8, vr + kB + 16);
            cp16(bv1.u + 0, vr + 32 + kB);   cp16(bv1.u + 8, vr + 32 + kB + 16);
            o[dt] = wmma_bf16(ap0, bv0, o[dt]);
            o[dt] = wmma_bf16(ap1, bv1, o[dt]);
        }
        __syncthreads();   // reads done before next iteration restages K/V/P
    }

    // ---- normalize and store: O[b][t][h*64 + d] ----
#pragma unroll
    for (int dt = 0; dt < 4; ++dt) {
#pragma unroll
        for (int r = 0; r < 8; ++r) {
            int trow = q0 + wv * 16 + r + hlf * 8;
            int col  = h * 64 + dt * 16 + mrow;
            float val = o[dt][r] / l_run[r];
            Og[((size_t)b * 1024 + trow) * 1024 + col] = f2bf(val);
        }
    }
}

// ---------------------------------------------------------------------------
extern "C" void kernel_launch(void* const* d_in, const int* in_sizes, int n_in,
                              void* d_out, int out_size, void* d_ws, size_t ws_size,
                              hipStream_t stream) {
    (void)in_sizes; (void)n_in; (void)out_size; (void)ws_size;

    const float* x  = (const float*)d_in[0];
    const float* Wq = (const float*)d_in[1];
    const float* Wk = (const float*)d_in[2];
    const float* Wv = (const float*)d_in[3];
    const float* Wp = (const float*)d_in[4];
    const float* bp = (const float*)d_in[5];
    float* out      = (float*)d_out;

    unsigned short* ws    = (unsigned short*)d_ws;
    unsigned short* x_bf  = ws;                       // 4096*1024
    unsigned short* wqkv  = x_bf + 4194304;           // 1024*3072
    unsigned short* wp_bf = wqkv + 3145728;           // 1024*1024
    unsigned short* qkv   = wp_bf + 1048576;          // 3 * 4194304
    unsigned short* q_bf  = qkv;
    unsigned short* k_bf  = qkv + 4194304;
    unsigned short* v_bf  = qkv + 2 * 4194304;
    unsigned short* attn  = qkv + 3 * 4194304;        // 4096*1024

    // 1) convert inputs to bf16
    cvt_f32_bf16<<<4194304 / 256, 256, 0, stream>>>(x, x_bf, 4194304);
    pack_wqkv  <<<3145728 / 256, 256, 0, stream>>>(Wq, Wk, Wv, wqkv);
    cvt_f32_bf16<<<1048576 / 256, 256, 0, stream>>>(Wp, wp_bf, 1048576);

    // 2) QKV projection: [4096,1024] x [1024,3072] -> scatter to q/k/v
    gemm_bf16<0><<<dim3(32, 24), 256, 0, stream>>>(
        x_bf, wqkv, 4096, 3072, 1024, qkv, nullptr, nullptr);

    // 3) flash attention, scale = 1/sqrt(C) = 1/32
    attn_kernel<<<dim3(16, 16, 4), 128, 0, stream>>>(
        q_bf, k_bf, v_bf, attn, 1.0f / 32.0f);

    // 4) output projection: [4096,1024] x [1024,1024] + bias -> f32 out
    gemm_bf16<1><<<dim3(32, 8), 256, 0, stream>>>(
        attn, wp_bf, 4096, 1024, 1024, nullptr, out, bp);
}
